// MultiHeadAttention_12919261626760
// MI455X (gfx1250) — matrix operand order repaired and checked by reading, not compiled, not run on hardware
//
#include <hip/hip_runtime.h>
#include <hip/hip_bf16.h>
#include <math.h>

#define D_MODEL 1024
#define NUM_HEADS 16
#define DEPTH 64
#define BATCH 8
#define SEQ 1024

typedef _Float16 v16h __attribute__((ext_vector_type(16)));
typedef _Float16 v8h  __attribute__((ext_vector_type(8)));
typedef _Float16 v4h  __attribute__((ext_vector_type(4)));
typedef float    v8f  __attribute__((ext_vector_type(8)));
typedef float    v4f  __attribute__((ext_vector_type(4)));

// ---- WMMA helper: D = A(16x32 f16) * B(32x16 f16) + C(16x16 f32) ----
static __device__ inline v8f wmma16(v16h a, v16h b, v8f c) {
  return __builtin_amdgcn_wmma_f32_16x16x32_f16(
      /*neg_a=*/false, a, /*neg_b=*/false, b,
      /*c_mod=*/(short)0, c, /*reuse_a=*/false, /*reuse_b=*/false);
}

// Fragment from 16 contiguous f16 (single 32B load).
static __device__ inline v16h frag_contig(const _Float16* p) {
  return *(const v16h*)p;
}

// Fragment from two runs of 8 contiguous f16 (A-matrix layout).
static __device__ inline v16h frag_split(const _Float16* p0, const _Float16* p1) {
  v8h lo = *(const v8h*)p0;
  v8h hi = *(const v8h*)p1;
  v16h r;
#pragma unroll
  for (int i = 0; i < 8; ++i) { r[i] = lo[i]; r[8 + i] = hi[i]; }
  return r;
}

// Fragment from two runs of 8 contiguous f32 (LDS probs -> f16).
static __device__ inline v16h frag_from_f32(const float* p0, const float* p1) {
  v4f a = *(const v4f*)p0;
  v4f b = *(const v4f*)(p0 + 4);
  v4f c = *(const v4f*)p1;
  v4f d = *(const v4f*)(p1 + 4);
  v16h r;
#pragma unroll
  for (int i = 0; i < 4; ++i) {
    r[i]      = (_Float16)a[i];
    r[4 + i]  = (_Float16)b[i];
    r[8 + i]  = (_Float16)c[i];
    r[12 + i] = (_Float16)d[i];
  }
  return r;
}

static __device__ inline float wred_max(float v) {
#pragma unroll
  for (int off = 16; off > 0; off >>= 1) v = fmaxf(v, __shfl_xor(v, off, 32));
  return v;
}
static __device__ inline float wred_sum(float v) {
#pragma unroll
  for (int off = 16; off > 0; off >>= 1) v += __shfl_xor(v, off, 32);
  return v;
}

// ---------------------------------------------------------------------------
// f32 -> f16 bulk conversion (vectorized: b128 in, b64 out).
// ---------------------------------------------------------------------------
__global__ __launch_bounds__(256) void cvt_kernel(
    const float* __restrict__ in, _Float16* __restrict__ out, int n4) {
  const int i = blockIdx.x * 256 + threadIdx.x;
  if (i < n4) {
    v4f x = ((const v4f*)in)[i];
    v4h y;
#pragma unroll
    for (int j = 0; j < 4; ++j) y[j] = (_Float16)x[j];
    ((v4h*)out)[i] = y;
  }
}

// ---------------------------------------------------------------------------
// Projection GEMM (all-f16 operands): out = X[M,K] @ W[N,K]^T + bias (f32).
// mode 0: split-head [B,H,S,DEPTH]   (Q, K)
// mode 1: split-head transposed [B,H,DEPTH,S]   (V)
// Block: 4 waves x 32 lanes; each wave owns a 16-row x 256-col output strip.
// ---------------------------------------------------------------------------
__global__ __launch_bounds__(128) void proj_kernel(
    const _Float16* __restrict__ X, const _Float16* __restrict__ W,
    const float* __restrict__ bias, _Float16* __restrict__ out, int mode) {
  const int lane = threadIdx.x;
  const int wave = threadIdx.y;
  const int half = lane >> 4;
  const int lm   = lane & 15;

  const int m0 = blockIdx.x * 64 + wave * 16;
  const int n0 = blockIdx.y * 256;

  v8f acc[16] = {};
  const _Float16* xrow = X + (size_t)(m0 + lm) * D_MODEL;

  for (int kc = 0; kc < D_MODEL; kc += 32) {
    v16h a = frag_split(xrow + kc + half * 16, xrow + kc + half * 16 + 8);
    v16h b[16];
#pragma unroll
    for (int t = 0; t < 16; ++t)  // 16 independent 32B loads -> one clause
      b[t] = frag_contig(W + (size_t)(n0 + t * 16 + lm) * D_MODEL + kc +
                         half * 16);
#pragma unroll
    for (int t = 0; t < 16; ++t) acc[t] = wmma16(a, b[t], acc[t]);
  }

  const int bb    = m0 / SEQ;
  const int sbase = (m0 % SEQ) + 8 * half;
#pragma unroll
  for (int t = 0; t < 16; ++t) {
    const int n  = n0 + t * 16 + lm;
    const float bv = bias[n];
    const int hh = n >> 6;
    const int dd = n & 63;
#pragma unroll
    for (int v = 0; v < 8; ++v) {
      const int s = sbase + v;
      const float val = acc[t][v] + bv;
      size_t idx;
      if (mode == 0)
        idx = ((size_t)((bb * NUM_HEADS + hh) * SEQ + s)) * DEPTH + dd;
      else
        idx = ((size_t)((bb * NUM_HEADS + hh) * DEPTH + dd)) * SEQ + s;
      out[idx] = (_Float16)val;
    }
  }
}

// ---------------------------------------------------------------------------
// Attention: one 16-query tile per block, 2 waves split the 1024 keys.
// scores -> mask -> softmax (LDS) -> write attn (f32) -> P@V (WMMA).
// ---------------------------------------------------------------------------
__global__ __launch_bounds__(64) void attn_kernel(
    const _Float16* __restrict__ Qh, const _Float16* __restrict__ Kh,
    const _Float16* __restrict__ Vt, const int* __restrict__ mask,
    float* __restrict__ attn_out, _Float16* __restrict__ O) {
  __shared__ float sc[16][1028];  // padded stride: conflict-free b128 reads

  const int lane = threadIdx.x;
  const int wave = threadIdx.y;  // 0..1
  const int half = lane >> 4;
  const int lm   = lane & 15;

  const int tile = blockIdx.x;
  const int qt = tile & 63;
  const int hh = (tile >> 6) & 15;
  const int bb = tile >> 10;
  const int q0 = qt * 16;

  const size_t headoff = (size_t)(bb * NUM_HEADS + hh) * SEQ * DEPTH;
  const _Float16* Qbase = Qh + headoff;
  const _Float16* Kbase = Kh + headoff;
  const _Float16* Vbase = Vt + headoff;  // [DEPTH][SEQ]
  const float scale = 1.0f / (8.0f + 1e-8f);

  // Q fragments for depth chunks [0,32) and [32,64) — loaded once.
  const _Float16* qrow = Qbase + (size_t)(q0 + lm) * DEPTH;
  const v16h aq0 = frag_split(qrow + half * 16, qrow + half * 16 + 8);
  const v16h aq1 = frag_split(qrow + 32 + half * 16, qrow + 40 + half * 16);

  // ---- phase 1: scores for this wave's half of the keys ----
#pragma unroll 2
  for (int kt = 0; kt < 32; ++kt) {
    const int kk  = wave * 512 + kt * 16;
    const int key = kk + lm;
    const _Float16* krow = Kbase + (size_t)key * DEPTH;
    v16h b0 = frag_contig(krow + half * 16);
    v16h b1 = frag_contig(krow + 32 + half * 16);
    v8f acc = {};
    acc = wmma16(aq0, b0, acc);
    acc = wmma16(aq1, b1, acc);
    const int mk = (key == 0) ? 1 : mask[bb * (SEQ - 1) + key - 1];
#pragma unroll
    for (int v = 0; v < 8; ++v) {
      const float x = acc[v] * scale;
      sc[v + 8 * half][key] = mk ? x : -INFINITY;
    }
  }
  __syncthreads();

  // ---- phase 2: softmax per row; wave w owns rows [8w, 8w+8) ----
  const size_t arow_base = ((size_t)(bb * NUM_HEADS + hh) * SEQ + q0) * SEQ;
  for (int rr = 0; rr < 8; ++rr) {
    const int r = wave * 8 + rr;
    float mx = -INFINITY;
    for (int c = lane; c < SEQ; c += 32) mx = fmaxf(mx, sc[r][c]);
    mx = wred_max(mx);
    float ssum = 0.f;
    for (int c = lane; c < SEQ; c += 32) {
      const float e = __expf(sc[r][c] - mx);
      sc[r][c] = e;
      ssum += e;
    }
    ssum = wred_sum(ssum);
    const float inv = 1.0f / ssum;
    float* arow = attn_out + arow_base + (size_t)r * SEQ;
    for (int c = lane; c < SEQ; c += 32) {
      const float p = sc[r][c] * inv;
      sc[r][c] = p;
      arow[c]  = p;
    }
  }
  __syncthreads();

  // ---- phase 3: O(16x64) = P(16x1024) @ V(1024x64); wave w owns 32 cols ----
  v8f oa0 = {};
  v8f oa1 = {};
#pragma unroll 2
  for (int kc = 0; kc < SEQ; kc += 32) {
    const float* prow = &sc[lm][0];
    v16h ap = frag_from_f32(prow + kc + half * 16, prow + kc + half * 16 + 8);
    const int d0 = wave * 32 + lm;
    v16h b0 = frag_contig(Vbase + (size_t)d0 * SEQ + kc + half * 16);
    v16h b1 = frag_contig(Vbase + (size_t)(d0 + 16) * SEQ + kc + half * 16);
    oa0 = wmma16(ap, b0, oa0);
    oa1 = wmma16(ap, b1, oa1);
  }

  const size_t obase = ((size_t)bb * SEQ + q0) * D_MODEL + hh * DEPTH;
#pragma unroll
  for (int v = 0; v < 8; ++v) {
    const int s_off = v + 8 * half;
    O[obase + (size_t)s_off * D_MODEL + wave * 32 + lm]      = (_Float16)oa0[v];
    O[obase + (size_t)s_off * D_MODEL + wave * 32 + 16 + lm] = (_Float16)oa1[v];
  }
}

// ---------------------------------------------------------------------------
// Dense GEMM (all f16 operands): out = X[M,K] @ W[N,K]^T + bias, f32 output.
// ---------------------------------------------------------------------------
__global__ __launch_bounds__(128) void dense_kernel(
    const _Float16* __restrict__ X, const _Float16* __restrict__ W,
    const float* __restrict__ bias, float* __restrict__ out) {
  const int lane = threadIdx.x;
  const int wave = threadIdx.y;
  const int half = lane >> 4;
  const int lm   = lane & 15;

  const int m0 = blockIdx.x * 64 + wave * 16;
  const int n0 = blockIdx.y * 256;

  v8f acc[16] = {};
  const _Float16* xrow = X + (size_t)(m0 + lm) * D_MODEL;

  for (int kc = 0; kc < D_MODEL; kc += 32) {
    v16h a = frag_split(xrow + kc + half * 16, xrow + kc + half * 16 + 8);
    v16h b[16];
#pragma unroll
    for (int t = 0; t < 16; ++t)
      b[t] = frag_contig(W + (size_t)(n0 + t * 16 + lm) * D_MODEL + kc +
                         half * 16);
#pragma unroll
    for (int t = 0; t < 16; ++t) acc[t] = wmma16(a, b[t], acc[t]);
  }

#pragma unroll
  for (int t = 0; t < 16; ++t) {
    const int n = n0 + t * 16 + lm;
    const float bv = bias[n];
#pragma unroll
    for (int v = 0; v < 8; ++v)
      out[(size_t)(m0 + v + 8 * half) * D_MODEL + n] = acc[t][v] + bv;
  }
}

// ---------------------------------------------------------------------------
extern "C" void kernel_launch(void* const* d_in, const int* in_sizes, int n_in,
                              void* d_out, int out_size, void* d_ws,
                              size_t ws_size, hipStream_t stream) {
  const float* q    = (const float*)d_in[0];
  const float* k    = (const float*)d_in[1];
  const float* v    = (const float*)d_in[2];
  const int*   mask = (const int*)d_in[3];
  const float* wq_w = (const float*)d_in[4];
  const float* wq_b = (const float*)d_in[5];
  const float* wk_w = (const float*)d_in[6];
  const float* wk_b = (const float*)d_in[7];
  const float* wv_w = (const float*)d_in[8];
  const float* wv_b = (const float*)d_in[9];
  const float* dw   = (const float*)d_in[10];
  const float* db   = (const float*)d_in[11];

  float* out  = (float*)d_out;                       // [B,S,D]   8M f32
  float* attn = out + (size_t)BATCH * SEQ * D_MODEL; // [B,H,S,S] 128M f32

  const size_t NTOK = (size_t)BATCH * SEQ * D_MODEL;       // 8M elements
  const size_t NW   = (size_t)D_MODEL * D_MODEL;           // 1M elements

  // Workspace (f16): projections + merged output + weight copies = 72 MB.
  _Float16* Qh  = (_Float16*)d_ws;   // [B,H,S,DEPTH]
  _Float16* Kh  = Qh + NTOK;         // [B,H,S,DEPTH]
  _Float16* Vt  = Kh + NTOK;         // [B,H,DEPTH,S]
  _Float16* O   = Vt + NTOK;         // [B,S,D] head-merged
  _Float16* Wqf = O + NTOK;
  _Float16* Wkf = Wqf + NW;
  _Float16* Wvf = Wkf + NW;
  _Float16* Wdf = Wvf + NW;

  // Transient f16 activations live in the (not-yet-written) attn region;
  // attn_kernel overwrites the whole region afterwards with real values.
  _Float16* Xq = (_Float16*)attn;
  _Float16* Xk = Xq + NTOK;
  _Float16* Xv = Xk + NTOK;

  // --- stage 0: f32 -> f16 conversions ---
  const int n4a = (int)(NTOK / 4);   // activations: 2M vec4
  const int n4w = (int)(NW / 4);     // weights:   256K vec4
  cvt_kernel<<<dim3((n4a + 255) / 256), dim3(256), 0, stream>>>(q, Xq, n4a);
  cvt_kernel<<<dim3((n4a + 255) / 256), dim3(256), 0, stream>>>(k, Xk, n4a);
  cvt_kernel<<<dim3((n4a + 255) / 256), dim3(256), 0, stream>>>(v, Xv, n4a);
  cvt_kernel<<<dim3((n4w + 255) / 256), dim3(256), 0, stream>>>(wq_w, Wqf, n4w);
  cvt_kernel<<<dim3((n4w + 255) / 256), dim3(256), 0, stream>>>(wk_w, Wkf, n4w);
  cvt_kernel<<<dim3((n4w + 255) / 256), dim3(256), 0, stream>>>(wv_w, Wvf, n4w);
  cvt_kernel<<<dim3((n4w + 255) / 256), dim3(256), 0, stream>>>(dw, Wdf, n4w);

  // --- stage 1: Q/K/V projections ---
  dim3 gblk(32, 4);
  dim3 ggrid(BATCH * SEQ / 64, D_MODEL / 256);
  proj_kernel<<<ggrid, gblk, 0, stream>>>(Xq, Wqf, wq_b, Qh, 0);
  proj_kernel<<<ggrid, gblk, 0, stream>>>(Xk, Wkf, wk_b, Kh, 0);
  proj_kernel<<<ggrid, gblk, 0, stream>>>(Xv, Wvf, wv_b, Vt, 1);

  // --- stage 2: attention (scores, softmax, attn output, P@V) ---
  attn_kernel<<<dim3(BATCH * NUM_HEADS * SEQ / 16), dim3(32, 2), 0, stream>>>(
      Qh, Kh, Vt, mask, attn, O);

  // --- stage 3: output projection ---
  dense_kernel<<<ggrid, gblk, 0, stream>>>(O, Wdf, db, out);
}
